// AutoEncoderFactorizedAttentionBlock_84439057039869
// MI455X (gfx1250) — matrix operand order repaired and checked by reading, not compiled, not run on hardware
//
#include <hip/hip_runtime.h>
#include <hip/hip_bf16.h>
#include <math.h>

#define N_LANE 20000
#define N_AGENT 10000
#define E_L2L 160000
#define E_A2A 100000
#define E_L2A 200000
#define LH 256
#define LFF 1024
#define LHEADS 8
#define AH 128
#define AFF 512
#define AHEADS 8
#define CH 64

#define ACT_NONE 0
#define ACT_RELU 1
#define ACT_SIG 2

// max K*M over all GEMMs (ff1 of l2l: 256x1024) -> packed dwords = K*M/2
#define WPACK_DWORDS (256 * 1024 / 2)

typedef __attribute__((ext_vector_type(16))) __bf16    bf16x16;
typedef __attribute__((ext_vector_type(8)))  float     floatx8;
typedef __attribute__((ext_vector_type(8)))  unsigned  uintx8;

// ---------------------------------------------------------------- utilities

__device__ __forceinline__ __bf16 f2bf(float f) { return (__bf16)f; }

// pack two f32 into one dword of bf16 {hi,lo}
__device__ __forceinline__ unsigned pack2bf(float lo, float hi) {
  union { __bf16 b; unsigned short s; } a, b;
  a.b = (__bf16)lo; b.b = (__bf16)hi;
  return ((unsigned)b.s << 16) | (unsigned)a.s;
}

__device__ __forceinline__ void atomicMaxFloat(float* addr, float val) {
  unsigned int* ua = (unsigned int*)addr;
  unsigned int cur = *ua;
  while (__uint_as_float(cur) < val) {
    unsigned int prev = atomicCAS(ua, cur, __float_as_uint(val));
    if (prev == cur) break;
    cur = prev;
  }
}

// ---------------------------------------------------------------- weight pack
// W[K,M] f32 row-major  ->  B-fragment-ordered bf16 pairs.
// Tile = 32 K x 16 M. Packed dword index:
//   P[((kc*(M/16) + mt)*16 + c)*16 + j] = {W[kc*32+2j+1, mt*16+c], W[kc*32+2j, mt*16+c]}
// A wave32 lane (col, hiHalf) then reads its 8 contiguous dwords at
//   tileBase + col*16 + hiHalf*8   (two global_load_b128).
// REQUIRES K % 32 == 0 and M % 16 == 0.

__global__ void pack_w_kernel(const float* __restrict__ W, unsigned* __restrict__ P,
                              int K, int M) {
  const int mtiles = M >> 4;
  size_t total = (size_t)(K >> 5) * mtiles * 256;
  size_t step  = (size_t)gridDim.x * blockDim.x;
  for (size_t i = (size_t)blockIdx.x * blockDim.x + threadIdx.x; i < total; i += step) {
    int j  = (int)(i & 15);
    int c  = (int)((i >> 4) & 15);
    int mt = (int)((i >> 8) % mtiles);
    int kc = (int)(i / ((size_t)mtiles << 8));
    int kk = kc * 32 + 2 * j;
    int mm = mt * 16 + c;
    P[i] = pack2bf(W[(size_t)kk * M + mm], W[(size_t)(kk + 1) * M + mm]);
  }
}

// ---------------------------------------------------------------- WMMA GEMM
// C[N,M] = act(A[N,K] @ W[K,M] + bias), W pre-packed by pack_w_kernel.
// REQUIRES K % 32 == 0 and M % 64 == 0 (all call sites satisfy this).
// 4 waves/block, each wave computes a 16-row x 64-col tile (4 WMMA accumulators
// sharing one A fragment). Block tile: 64 rows x 64 cols. No LDS, no barriers.

__global__ __launch_bounds__(128) void wmma_gemm_kernel(
    const float* __restrict__ A, const unsigned* __restrict__ WP,
    const float* __restrict__ bias, float* __restrict__ C,
    int N, int K, int M, int act) {
  const int tid    = threadIdx.x;
  const int wave   = tid >> 5;
  const int lane   = tid & 31;
  const int m0     = blockIdx.x * 64;
  const int n0     = blockIdx.y * 64 + wave * 16;
  const int col    = lane & 15;
  const int hiHalf = lane >> 4;          // 0 or 1
  int rowA = n0 + col;                   // this lane's A row (M of WMMA)
  if (rowA >= N) rowA = N - 1;           // clamp: garbage rows are never stored
  const float* __restrict__ arow = A + (size_t)rowA * K;
  const int kb     = hiHalf * 16;        // lanes 0-15: K 0..15; 16-31: K 16..31
  const int mtiles = M >> 4;
  const int mt0    = m0 >> 4;
  const int bunit  = col * 2 + hiHalf;   // uintx8 unit within a packed tile

  floatx8 acc0 = {}, acc1 = {}, acc2 = {}, acc3 = {};
  for (int k0 = 0, kc = 0; k0 < K; k0 += 32, ++kc) {
    // ---- A fragment: four contiguous float4 loads, native bf16 converts
    const float4* ap = (const float4*)(arow + k0 + kb);
    float4 a0 = ap[0], a1 = ap[1];       // K = kb .. kb+7
    float4 a2 = ap[2], a3 = ap[3];       // K = kb+8 .. kb+15
    __builtin_prefetch(arow + k0 + 32, 0, 1);
    bf16x16 afrag;
    afrag[0]  = f2bf(a0.x); afrag[1]  = f2bf(a0.y);
    afrag[2]  = f2bf(a0.z); afrag[3]  = f2bf(a0.w);
    afrag[4]  = f2bf(a1.x); afrag[5]  = f2bf(a1.y);
    afrag[6]  = f2bf(a1.z); afrag[7]  = f2bf(a1.w);
    afrag[8]  = f2bf(a2.x); afrag[9]  = f2bf(a2.y);
    afrag[10] = f2bf(a2.z); afrag[11] = f2bf(a2.w);
    afrag[12] = f2bf(a3.x); afrag[13] = f2bf(a3.y);
    afrag[14] = f2bf(a3.z); afrag[15] = f2bf(a3.w);

    // ---- 4 B fragments straight from the packed weights (L2-resident)
    const uintx8* bp = (const uintx8*)(WP + ((size_t)kc * mtiles + mt0) * 256);
    bf16x16 b0 = __builtin_bit_cast(bf16x16, bp[0 * 32 + bunit]);
    bf16x16 b1 = __builtin_bit_cast(bf16x16, bp[1 * 32 + bunit]);
    bf16x16 b2 = __builtin_bit_cast(bf16x16, bp[2 * 32 + bunit]);
    bf16x16 b3 = __builtin_bit_cast(bf16x16, bp[3 * 32 + bunit]);

    acc0 = __builtin_amdgcn_wmma_f32_16x16x32_bf16(false, afrag, false, b0,
                                                   (short)0, acc0, false, false);
    acc1 = __builtin_amdgcn_wmma_f32_16x16x32_bf16(false, afrag, false, b1,
                                                   (short)0, acc1, false, false);
    acc2 = __builtin_amdgcn_wmma_f32_16x16x32_bf16(false, afrag, false, b2,
                                                   (short)0, acc2, false, false);
    acc3 = __builtin_amdgcn_wmma_f32_16x16x32_bf16(false, afrag, false, b3,
                                                   (short)0, acc3, false, false);
  }

  // ---- epilogue; C/D layout: lane&15 = col, VGPR r = row r (+8 for hi half)
  const int rbase = n0 + hiHalf * 8;
#pragma unroll
  for (int t = 0; t < 4; ++t) {
    const floatx8& acc = (t == 0) ? acc0 : (t == 1) ? acc1 : (t == 2) ? acc2 : acc3;
    const int mcol = m0 + t * 16 + col;          // M % 64 == 0: always in range
    float bv = bias ? bias[mcol] : 0.0f;
#pragma unroll
    for (int r = 0; r < 8; ++r) {
      int row = rbase + r;
      if (row < N) {
        float v = acc[r] + bv;
        if (act == ACT_RELU)     v = fmaxf(v, 0.0f);
        else if (act == ACT_SIG) v = 1.0f / (1.0f + __expf(-v));
        C[(size_t)row * M + mcol] = v;
      }
    }
  }
}

// ---------------------------------------------------------------- LayerNorm
// one wave per row; optional residual add (y = res + ln(x)) and ReLU.

__global__ void ln_kernel(const float* __restrict__ x, const float* __restrict__ g,
                          const float* __restrict__ b, const float* __restrict__ res,
                          float* __restrict__ y, int N, int D, int act) {
  int wrow = (int)((blockIdx.x * (size_t)blockDim.x + threadIdx.x) >> 5);
  int lane = threadIdx.x & 31;
  if (wrow >= N) return;
  const float* row = x + (size_t)wrow * D;
  float sum = 0.f, sq = 0.f;
  for (int i = lane; i < D; i += 32) { float v = row[i]; sum += v; sq += v * v; }
#pragma unroll
  for (int off = 16; off; off >>= 1) {
    sum += __shfl_xor(sum, off, 32);
    sq  += __shfl_xor(sq,  off, 32);
  }
  float mean = sum / (float)D;
  float var  = sq / (float)D - mean * mean;
  float rstd = rsqrtf(var + 1e-5f);
  float* out = y + (size_t)wrow * D;
  for (int i = lane; i < D; i += 32) {
    float v = (row[i] - mean) * rstd * g[i] + b[i];
    if (res) v += res[(size_t)wrow * D + i];
    if (act) v = fmaxf(v, 0.0f);
    out[i] = v;
  }
}

// ---------------------------------------------------------------- edge kernels

__global__ void edge_sim_kernel(const float* __restrict__ q, const float* __restrict__ k,
                                const float* __restrict__ kr,
                                const int* __restrict__ src, const int* __restrict__ dst,
                                float* __restrict__ sim, float* __restrict__ smax,
                                int E, int H, int Dh, float scale) {
  size_t total = (size_t)E * H;
  size_t step  = (size_t)gridDim.x * blockDim.x;
  for (size_t i = (size_t)blockIdx.x * blockDim.x + threadIdx.x; i < total; i += step) {
    int e = (int)(i / H), h = (int)(i % H);
    int sN = src[e], dN = dst[e];
    const float* qp  = q + ((size_t)dN * H + h) * Dh;
    const float* kp  = k + ((size_t)sN * H + h) * Dh;
    const float* krp = kr ? kr + ((size_t)e * H + h) * Dh : nullptr;
    float acc = 0.f;
    for (int j = 0; j < Dh; ++j) acc += qp[j] * (kp[j] + (krp ? krp[j] : 0.f));
    acc *= scale;
    sim[i] = acc;
    atomicMaxFloat(&smax[(size_t)dN * H + h], acc);
  }
}

__global__ void edge_exp_kernel(float* __restrict__ sim, const float* __restrict__ smax,
                                float* __restrict__ ssum, const int* __restrict__ dst,
                                int E, int H) {
  size_t total = (size_t)E * H;
  size_t step  = (size_t)gridDim.x * blockDim.x;
  for (size_t i = (size_t)blockIdx.x * blockDim.x + threadIdx.x; i < total; i += step) {
    int e = (int)(i / H), h = (int)(i % H);
    int dN = dst[e];
    float w = __expf(sim[i] - smax[(size_t)dN * H + h]);
    sim[i] = w;
    atomicAdd(&ssum[(size_t)dN * H + h], w);
  }
}

__global__ void edge_agg_kernel(const float* __restrict__ v, const float* __restrict__ vr,
                                const float* __restrict__ w, const float* __restrict__ ssum,
                                const int* __restrict__ src, const int* __restrict__ dst,
                                float* __restrict__ agg, int E, int HD, int Dh, int H) {
  size_t total = (size_t)E * HD;
  size_t step  = (size_t)gridDim.x * blockDim.x;
  for (size_t i = (size_t)blockIdx.x * blockDim.x + threadIdx.x; i < total; i += step) {
    size_t e = i / HD; int j = (int)(i % HD); int h = j / Dh;
    int sN = src[e], dN = dst[e];
    float wt = w[e * H + h] / ssum[(size_t)dN * H + h];
    float vv = v[(size_t)sN * HD + j] + (vr ? vr[i] : 0.f);
    atomicAdd(&agg[(size_t)dN * HD + j], wt * vv);
  }
}

__global__ void edge_cat_kernel(const float* __restrict__ x, const int* __restrict__ src,
                                const int* __restrict__ dst, float* __restrict__ y,
                                int E, int D) {
  size_t total = (size_t)E * 2 * D;
  size_t step  = (size_t)gridDim.x * blockDim.x;
  for (size_t i = (size_t)blockIdx.x * blockDim.x + threadIdx.x; i < total; i += step) {
    size_t e = i / (2 * D); int j = (int)(i % (2 * D));
    y[i] = (j < D) ? x[(size_t)dst[e] * D + j] : x[(size_t)src[e] * D + (j - D)];
  }
}

// ---------------------------------------------------------------- elementwise

__global__ void fill_kernel(float* __restrict__ p, float v, size_t n) {
  size_t step = (size_t)gridDim.x * blockDim.x;
  for (size_t i = (size_t)blockIdx.x * blockDim.x + threadIdx.x; i < n; i += step) p[i] = v;
}

__global__ void gate_kernel(const float* __restrict__ agg, const float* __restrict__ g,
                            const float* __restrict__ s, float* __restrict__ y, size_t n) {
  size_t step = (size_t)gridDim.x * blockDim.x;
  for (size_t i = (size_t)blockIdx.x * blockDim.x + threadIdx.x; i < n; i += step)
    y[i] = agg[i] + g[i] * (s[i] - agg[i]);
}

__global__ void add_act_kernel(const float* __restrict__ a, const float* __restrict__ b,
                               float* __restrict__ y, size_t n, int act) {
  size_t step = (size_t)gridDim.x * blockDim.x;
  for (size_t i = (size_t)blockIdx.x * blockDim.x + threadIdx.x; i < n; i += step) {
    float v = a[i] + b[i];
    if (act) v = fmaxf(v, 0.0f);
    y[i] = v;
  }
}

__global__ void concat2_kernel(const float* __restrict__ a, const float* __restrict__ b,
                               float* __restrict__ y, int N, int D1, int D2) {
  size_t total = (size_t)N * (D1 + D2);
  size_t step  = (size_t)gridDim.x * blockDim.x;
  for (size_t i = (size_t)blockIdx.x * blockDim.x + threadIdx.x; i < total; i += step) {
    size_t row = i / (D1 + D2); int j = (int)(i % (D1 + D2));
    y[i] = (j < D1) ? a[row * D1 + j] : b[row * D2 + (j - D1)];
  }
}

// ---------------------------------------------------------------- host side

struct Lin  { const float* w; const float* b; };
struct LNp  { const float* g; const float* b; };
struct AttnP {
  LNp pre; Lin q, k, v, s, g, out; LNp post, ffpre; Lin ff1, ff2; LNp ffpost;
  LNp prer; Lin kr, vr;
};
struct RMLP { Lin l0; LNp n0; Lin l1; LNp n1; Lin tl; LNp tn; };

static inline int ngrid(size_t n, int blk) {
  size_t g = (n + (size_t)blk - 1) / (size_t)blk;
  if (g > 32768) g = 32768;
  if (g < 1) g = 1;
  return (int)g;
}

// pack weights then run the WMMA GEMM (stream-ordered; wp reused across calls)
static void gemm(hipStream_t st, unsigned* wp, const float* A, const float* W,
                 const float* b, float* C, int N, int K, int M, int act) {
  size_t pdw = (size_t)K * M / 2;
  pack_w_kernel<<<ngrid(pdw, 256), 256, 0, st>>>(W, wp, K, M);
  dim3 g((M + 63) / 64, (N + 63) / 64);
  wmma_gemm_kernel<<<g, 128, 0, st>>>(A, wp, b, C, N, K, M, act);
}

static void ln_launch(hipStream_t st, const float* x, const float* g, const float* b,
                      const float* res, float* y, int N, int D, int act) {
  ln_kernel<<<(N + 7) / 8, 256, 0, st>>>(x, g, b, res, y, N, D, act);
}

static float* arena_alloc(float* wsf, size_t& cur, size_t n) {
  float* p = wsf + cur;
  cur += (n + 255) & ~(size_t)255;
  return p;
}

static void run_attn(hipStream_t st, float* wsf, size_t& cur, unsigned* wp,
                     const float* x_in, float* x_out, int N, int hd,
                     int H, int Dh, int ff,
                     const float* r, int rdim, int E,
                     const int* src, const int* dst, const AttnP& p) {
  size_t mark = cur;
  const int HD = H * Dh;
  const float scale = 1.0f / sqrtf((float)Dh);

  float* xs = arena_alloc(wsf, cur, (size_t)N * hd);
  ln_launch(st, x_in, p.pre.g, p.pre.b, nullptr, xs, N, hd, 0);

  float* rln = nullptr;
  if (r) {
    rln = arena_alloc(wsf, cur, (size_t)E * rdim);
    ln_launch(st, r, p.prer.g, p.prer.b, nullptr, rln, E, rdim, 0);
  }

  float* q = arena_alloc(wsf, cur, (size_t)N * HD);
  float* k = arena_alloc(wsf, cur, (size_t)N * HD);
  float* v = arena_alloc(wsf, cur, (size_t)N * HD);
  gemm(st, wp, xs, p.q.w, p.q.b, q, N, hd, HD, ACT_NONE);
  gemm(st, wp, xs, p.k.w, nullptr, k, N, hd, HD, ACT_NONE);
  gemm(st, wp, xs, p.v.w, p.v.b, v, N, hd, HD, ACT_NONE);

  float* smax = arena_alloc(wsf, cur, (size_t)N * H);
  float* ssum = arena_alloc(wsf, cur, (size_t)N * H);
  fill_kernel<<<ngrid((size_t)N * H, 256), 256, 0, st>>>(smax, -1e30f, (size_t)N * H);
  fill_kernel<<<ngrid((size_t)N * H, 256), 256, 0, st>>>(ssum, 0.0f, (size_t)N * H);

  float* sim = arena_alloc(wsf, cur, (size_t)E * H);
  float* eproj = nullptr;
  if (r) {
    eproj = arena_alloc(wsf, cur, (size_t)E * HD);
    gemm(st, wp, rln, p.kr.w, nullptr, eproj, E, rdim, HD, ACT_NONE);  // k_r
  }
  edge_sim_kernel<<<ngrid((size_t)E * H, 256), 256, 0, st>>>(
      q, k, eproj, src, dst, sim, smax, E, H, Dh, scale);
  edge_exp_kernel<<<ngrid((size_t)E * H, 256), 256, 0, st>>>(sim, smax, ssum, dst, E, H);

  float* agg = arena_alloc(wsf, cur, (size_t)N * HD);
  fill_kernel<<<ngrid((size_t)N * HD, 256), 256, 0, st>>>(agg, 0.0f, (size_t)N * HD);
  if (r) gemm(st, wp, rln, p.vr.w, p.vr.b, eproj, E, rdim, HD, ACT_NONE);  // reuse as v_r
  edge_agg_kernel<<<ngrid((size_t)E * HD, 256), 256, 0, st>>>(
      v, eproj, sim, ssum, src, dst, agg, E, HD, Dh, H);

  // gate: g = sigmoid(lin(concat[agg, xd]))
  float* cat = arena_alloc(wsf, cur, (size_t)N * (HD + hd));
  concat2_kernel<<<ngrid((size_t)N * (HD + hd), 256), 256, 0, st>>>(agg, xs, cat, N, HD, hd);
  float* gbuf = arena_alloc(wsf, cur, (size_t)N * HD);
  gemm(st, wp, cat, p.g.w, p.g.b, gbuf, N, HD + hd, HD, ACT_SIG);
  float* sbuf = arena_alloc(wsf, cur, (size_t)N * HD);
  gemm(st, wp, xs, p.s.w, p.s.b, sbuf, N, hd, HD, ACT_NONE);

  // upd -> q (q is dead); attn_out -> k (k is dead); hd == HD for all layers here
  gate_kernel<<<ngrid((size_t)N * HD, 256), 256, 0, st>>>(agg, gbuf, sbuf, q, (size_t)N * HD);
  gemm(st, wp, q, p.out.w, p.out.b, k, N, HD, hd, ACT_NONE);

  // x_out = x_in + ln(attn_out)
  ln_launch(st, k, p.post.g, p.post.b, x_in, x_out, N, hd, 0);

  // FFN: x_out += ln(ff2(relu(ff1(ln(x_out)))))
  ln_launch(st, x_out, p.ffpre.g, p.ffpre.b, nullptr, xs, N, hd, 0);
  float* h1 = arena_alloc(wsf, cur, (size_t)N * ff);
  gemm(st, wp, xs, p.ff1.w, p.ff1.b, h1, N, hd, ff, ACT_RELU);
  gemm(st, wp, h1, p.ff2.w, p.ff2.b, k, N, ff, hd, ACT_NONE);
  ln_launch(st, k, p.ffpost.g, p.ffpost.b, x_out, x_out, N, hd, 0);

  cur = mark;
}

static void run_rmlp(hipStream_t st, float* wsf, size_t& cur, unsigned* wp,
                     const float* x, float* y, int N, int in, int h, const RMLP& m) {
  size_t mark = cur;
  float* b1 = arena_alloc(wsf, cur, (size_t)N * h);
  float* b2 = arena_alloc(wsf, cur, (size_t)N * h);
  float* b3 = arena_alloc(wsf, cur, (size_t)N * h);
  gemm(st, wp, x, m.l0.w, m.l0.b, b1, N, in, h, ACT_NONE);
  ln_launch(st, b1, m.n0.g, m.n0.b, nullptr, b1, N, h, 1);      // relu(ln(.))
  gemm(st, wp, b1, m.l1.w, m.l1.b, b2, N, h, h, ACT_NONE);
  ln_launch(st, b2, m.n1.g, m.n1.b, nullptr, b2, N, h, 0);
  gemm(st, wp, x, m.tl.w, m.tl.b, b3, N, in, h, ACT_NONE);
  ln_launch(st, b3, m.tn.g, m.tn.b, nullptr, b3, N, h, 0);
  add_act_kernel<<<ngrid((size_t)N * h, 256), 256, 0, st>>>(b2, b3, y, (size_t)N * h, 1);
  cur = mark;
}

extern "C" void kernel_launch(void* const* d_in, const int* in_sizes, int n_in,
                              void* d_out, int out_size, void* d_ws, size_t ws_size,
                              hipStream_t stream) {
  (void)in_sizes; (void)n_in; (void)out_size; (void)ws_size;

  const float* agent_x = (const float*)d_in[0];
  const float* lane_x  = (const float*)d_in[1];
  const float* conn_x  = (const float*)d_in[2];

  // params pytree, depth-first in dict insertion order (100 leaves)
  int pi = 3;
  auto nxt  = [&]() { return (const float*)d_in[pi++]; };
  auto rdLN = [&](LNp& l) { l.g = nxt(); l.b = nxt(); };
  auto rdLin = [&](Lin& l, bool bias) { l.w = nxt(); l.b = bias ? nxt() : nullptr; };
  auto rdAttn = [&](AttnP& a, bool pos) {
    rdLN(a.pre);
    rdLin(a.q, true); rdLin(a.k, false); rdLin(a.v, true);
    rdLin(a.s, true); rdLin(a.g, true);  rdLin(a.out, true);
    rdLN(a.post); rdLN(a.ffpre);
    rdLin(a.ff1, true); rdLin(a.ff2, true);
    rdLN(a.ffpost);
    if (pos) { rdLN(a.prer); rdLin(a.kr, false); rdLin(a.vr, true); }
    else { a.prer.g = a.prer.b = nullptr; a.kr.w = a.kr.b = nullptr; a.vr.w = a.vr.b = nullptr; }
  };
  AttnP l2l, l2a, a2a; Lin down; RMLP eun, eue;
  rdAttn(l2l, true);
  rdAttn(l2a, false);
  rdAttn(a2a, false);
  rdLin(down, true);
  auto rdR = [&](RMLP& m) {
    rdLin(m.l0, true); rdLN(m.n0); rdLin(m.l1, true); rdLN(m.n1); rdLin(m.tl, true); rdLN(m.tn);
  };
  rdR(eun); rdR(eue);

  const int* a2a_src = (const int*)d_in[pi]; const int* a2a_dst = a2a_src + E_A2A; pi++;
  const int* l2l_src = (const int*)d_in[pi]; const int* l2l_dst = l2l_src + E_L2L; pi++;
  const int* l2a_src = (const int*)d_in[pi]; const int* l2a_dst = l2a_src + E_L2A; pi++;

  // output layout: agent (10000x128) ++ lane (20000x256) ++ conn (160000x64)
  float* out_agent = (float*)d_out;
  float* out_lane  = out_agent + (size_t)N_AGENT * AH;
  float* out_conn  = out_lane + (size_t)N_LANE * LH;

  float* wsf = (float*)d_ws;
  size_t cur = 0;
  unsigned* wpack = (unsigned*)arena_alloc(wsf, cur, WPACK_DWORDS);
  float* lane_ds  = arena_alloc(wsf, cur, (size_t)N_LANE * AH);
  float* ad       = arena_alloc(wsf, cur, (size_t)(N_LANE + N_AGENT) * AH);
  size_t mark = cur;

  // 1) lane = l2l attention (with positional edge features) -> out_lane
  run_attn(stream, wsf, cur, wpack, lane_x, out_lane, N_LANE, LH, LHEADS, LH / LHEADS,
           LFF, conn_x, CH, E_L2L, l2l_src, l2l_dst, l2l);
  cur = mark;

  // 2) lane_ds = down(lane); ad = concat(lane_ds, agent_x)
  gemm(stream, wpack, out_lane, down.w, down.b, lane_ds, N_LANE, LH, AH, ACT_NONE);
  hipMemcpyAsync(ad, lane_ds, sizeof(float) * (size_t)N_LANE * AH,
                 hipMemcpyDeviceToDevice, stream);
  hipMemcpyAsync(ad + (size_t)N_LANE * AH, agent_x, sizeof(float) * (size_t)N_AGENT * AH,
                 hipMemcpyDeviceToDevice, stream);

  // 3) l2a attention over ad (dst indices already offset by N_LANE), in place
  run_attn(stream, wsf, cur, wpack, ad, ad, N_LANE + N_AGENT, AH, AHEADS, AH / AHEADS,
           AFF, nullptr, 0, E_L2A, l2a_src, l2a_dst, l2a);
  cur = mark;

  // 4) a2a attention over agent slice -> out_agent
  run_attn(stream, wsf, cur, wpack, ad + (size_t)N_LANE * AH, out_agent, N_AGENT, AH,
           AHEADS, AH / AHEADS, AFF, nullptr, 0, E_A2A, a2a_src, a2a_dst, a2a);
  cur = mark;

  // 5) edge update: nf = [lane_ds[dst] | lane_ds[src]]; un = rmlp(nf);
  //    conn = rmlp([un | lane_conn])
  float* nf = arena_alloc(wsf, cur, (size_t)E_L2L * 2 * AH);
  edge_cat_kernel<<<ngrid((size_t)E_L2L * 2 * AH, 256), 256, 0, stream>>>(
      lane_ds, l2l_src, l2l_dst, nf, E_L2L, AH);
  float* un = arena_alloc(wsf, cur, (size_t)E_L2L * CH);
  run_rmlp(stream, wsf, cur, wpack, nf, un, E_L2L, 2 * AH, CH, eun);
  float* ec = arena_alloc(wsf, cur, (size_t)E_L2L * 2 * CH);
  concat2_kernel<<<ngrid((size_t)E_L2L * 2 * CH, 256), 256, 0, stream>>>(
      un, conn_x, ec, E_L2L, CH, CH);
  run_rmlp(stream, wsf, cur, wpack, ec, out_conn, E_L2L, 2 * CH, CH, eue);
}
